// FeatureMessagePassingModel_23201413333127
// MI455X (gfx1250) — matrix operand order repaired and checked by reading, not compiled, not run on hardware
//
#include <hip/hip_runtime.h>
#include <hip/hip_bf16.h>

typedef __attribute__((ext_vector_type(16))) __bf16 v16bf;
typedef __attribute__((ext_vector_type(8)))  __bf16 v8bf;
typedef __attribute__((ext_vector_type(8)))  float  v8f;

#define FDIM 128

// ---------------------------------------------------------------------------
// Small utility kernels
// ---------------------------------------------------------------------------
__global__ void cvt_w_kernel(const float* __restrict__ w, __bf16* __restrict__ o, int n) {
    int i = blockIdx.x * blockDim.x + threadIdx.x;
    if (i < n) o[i] = (__bf16)w[i];
}

__global__ void zero_int_kernel(int* __restrict__ p, int n) {
    int i = blockIdx.x * blockDim.x + threadIdx.x;
    if (i < n) p[i] = 0;
}

__global__ void edge_degree_kernel(const int* __restrict__ dst, int* __restrict__ deg, int E) {
    int e = blockIdx.x * blockDim.x + threadIdx.x;
    if (e < E) atomicAdd(&deg[dst[e]], 1);
}

__global__ void rsqrt_kernel(const int* __restrict__ deg, float* __restrict__ dinv, int n) {
    int i = blockIdx.x * blockDim.x + threadIdx.x;
    if (i < n) dinv[i] = rsqrtf((float)deg[i] + 1.0f);   // +1 for self-loop
}

// out[i][:] = dinv[i]^2 * h1[i][:]   (self-loop contribution, no atomics needed)
__global__ void self_init_kernel(const float* __restrict__ h1, const float* __restrict__ dinv,
                                 float* __restrict__ out, int N) {
    int t = blockIdx.x * blockDim.x + threadIdx.x;
    int node = t >> 5, lane = t & 31;
    if (node >= N) return;
    float s = dinv[node];
    s = s * s;
    const float4 v = *(const float4*)(h1 + (size_t)node * FDIM + lane * 4);
    float4 r = make_float4(v.x * s, v.y * s, v.z * s, v.w * s);
    *(float4*)(out + (size_t)node * FDIM + lane * 4) = r;
}

// one wave per edge; lanes cover 128 features as float4; f32 hardware atomics
__global__ void edge_scatter_kernel(const float* __restrict__ h1, const int* __restrict__ src,
                                    const int* __restrict__ dst, const float* __restrict__ dinv,
                                    float* __restrict__ out, int E) {
    int t = blockIdx.x * blockDim.x + threadIdx.x;
    int e = t >> 5, lane = t & 31;
    if (e >= E) return;
    const int s = src[e];
    const int d = dst[e];
    const float norm = dinv[s] * dinv[d];
    const float4 v = *(const float4*)(h1 + (size_t)s * FDIM + lane * 4);
    float* o = out + (size_t)d * FDIM + lane * 4;
    unsafeAtomicAdd(o + 0, v.x * norm);
    unsafeAtomicAdd(o + 1, v.y * norm);
    unsafeAtomicAdd(o + 2, v.z * norm);
    unsafeAtomicAdd(o + 3, v.w * norm);
}

__global__ void bias_relu_kernel(float* __restrict__ h, const float* __restrict__ bias, int N) {
    int t = blockIdx.x * blockDim.x + threadIdx.x;
    int node = t >> 5, lane = t & 31;
    if (node >= N) return;
    float* p = h + (size_t)node * FDIM + lane * 4;
    float4 v = *(const float4*)p;
    const float4 b = *(const float4*)(bias + lane * 4);
    v.x = fmaxf(v.x + b.x, 0.0f);
    v.y = fmaxf(v.y + b.y, 0.0f);
    v.z = fmaxf(v.z + b.z, 0.0f);
    v.w = fmaxf(v.w + b.w, 0.0f);
    *(float4*)p = v;
}

// ---------------------------------------------------------------------------
// WMMA GEMM:  C[M x Ncols] = A[M x 128] * W[128 x Ncols] (+bias)
// A fp32 row-major, W bf16 row-major, C fp32 row-major. NT = ceil(Ncols/16).
// One wave computes a 16-row strip across ALL NT column tiles: A fragment is
// loaded once per K-step and reused for NT independent v_wmma ops.
// W staged TRANSPOSED in LDS ([col][k], padded stride 136 -> lanes hit banks
// 4 apart, conflict-free); each B fragment = two aligned ds_load_b128.
// B fragments are software-pipelined one tile ahead so the next ds_load
// overlaps the current v_wmma instead of stalling on s_wait_dscnt.
// ---------------------------------------------------------------------------
template<int NT>
__global__ void __launch_bounds__(256)
gcn_gemm_wmma(const float* __restrict__ A, const __bf16* __restrict__ Wb,
              const float* __restrict__ bias, float* __restrict__ Cout,
              int M, int Ncols) {
    constexpr int COLS = NT * 16;
    constexpr int LDSK = FDIM + 8;            // padded k-stride (272 B)
    __shared__ __bf16 Wt[COLS * LDSK];        // W transposed: [col][k]

    for (int idx = threadIdx.x; idx < COLS * FDIM; idx += 256) {
        const int col = idx % COLS;           // fastest -> coalesced global read
        const int k   = idx / COLS;
        Wt[col * LDSK + k] = (col < Ncols) ? Wb[k * Ncols + col] : (__bf16)0.0f;
    }
    __syncthreads();

    const int wave = threadIdx.x >> 5;
    const int lane = threadIdx.x & 31;
    const int tile_m = blockIdx.x * 8 + wave;
    const int tiles_m = (M + 15) >> 4;
    if (tile_m >= tiles_m) return;

    const int half = lane >> 4;
    const int l15  = lane & 15;

    int row = tile_m * 16 + l15;
    if (row >= M) row = M - 1;                // clamp (store is guarded)
    const float* arow = A + (size_t)row * FDIM;

    v8f acc[NT];
#pragma unroll
    for (int nt = 0; nt < NT; ++nt) acc[nt] = (v8f){};

#pragma unroll
    for (int k0 = 0; k0 < FDIM; k0 += 32) {
        // ---- A fragment: 16-bit A 16x32 layout --------------------------
        // lane<16:  K=k0+0..7 then k0+16..23 ; lane>=16: +8
        const int ka = k0 + half * 8;
        const float4 f0 = *(const float4*)(arow + ka);
        const float4 f1 = *(const float4*)(arow + ka + 4);
        const float4 f2 = *(const float4*)(arow + ka + 16);
        const float4 f3 = *(const float4*)(arow + ka + 20);
        v16bf a;
        a[0]  = (__bf16)f0.x; a[1]  = (__bf16)f0.y; a[2]  = (__bf16)f0.z; a[3]  = (__bf16)f0.w;
        a[4]  = (__bf16)f1.x; a[5]  = (__bf16)f1.y; a[6]  = (__bf16)f1.z; a[7]  = (__bf16)f1.w;
        a[8]  = (__bf16)f2.x; a[9]  = (__bf16)f2.y; a[10] = (__bf16)f2.z; a[11] = (__bf16)f2.w;
        a[12] = (__bf16)f3.x; a[13] = (__bf16)f3.y; a[14] = (__bf16)f3.z; a[15] = (__bf16)f3.w;

        // ---- B fragments: lane l15 = output col, K=kb..kb+7 then kb+16..kb+23 ------
        const int kb = k0 + half * 8;
        auto loadB = [&](int nt) -> v16bf {
            const __bf16* bp = &Wt[(nt * 16 + l15) * LDSK + kb];
            const v8bf b0 = *(const v8bf*)bp;          // 16B-aligned ds_load_b128
            const v8bf b1 = *(const v8bf*)(bp + 16);
            return __builtin_shufflevector(
                b0, b1, 0, 1, 2, 3, 4, 5, 6, 7, 8, 9, 10, 11, 12, 13, 14, 15);
        };

        v16bf bcur = loadB(0);
#pragma unroll
        for (int nt = 0; nt < NT; ++nt) {
            v16bf bnext = bcur;
            if (nt + 1 < NT) bnext = loadB(nt + 1);    // prefetch next tile's B
            acc[nt] = __builtin_amdgcn_wmma_f32_16x16x32_bf16(
                /*neg_a=*/false, a, /*neg_b=*/false, bcur,
                /*c_mod=*/(short)0, acc[nt], /*reuse_a=*/false, /*reuse_b=*/false);
            bcur = bnext;
        }
    }

    // ---- store: C/D layout: VGPR i -> row tile_m*16 + half*8 + i, col l15
#pragma unroll
    for (int nt = 0; nt < NT; ++nt) {
        const int gcol = nt * 16 + l15;
        if (gcol < Ncols) {
            const float bv = bias ? bias[gcol] : 0.0f;
#pragma unroll
            for (int i = 0; i < 8; ++i) {
                const int grow = tile_m * 16 + half * 8 + i;
                if (grow < M) Cout[(size_t)grow * Ncols + gcol] = acc[nt][i] + bv;
            }
        }
    }
}

// ---------------------------------------------------------------------------
// Host-side orchestration
// ---------------------------------------------------------------------------
static inline size_t align_up(size_t x, size_t a) { return (x + a - 1) & ~(a - 1); }

extern "C" void kernel_launch(void* const* d_in, const int* in_sizes, int n_in,
                              void* d_out, int out_size, void* d_ws, size_t ws_size,
                              hipStream_t stream) {
    const float* x       = (const float*)d_in[0];
    const int*   ei      = (const int*)d_in[1];
    const float* pre_W   = (const float*)d_in[2];
    const float* pre_b   = (const float*)d_in[3];
    const float* conv1_W = (const float*)d_in[4];
    const float* conv1_b = (const float*)d_in[5];
    const float* conv2_W = (const float*)d_in[6];
    const float* conv2_b = (const float*)d_in[7];
    const float* post_W  = (const float*)d_in[8];
    const float* post_b  = (const float*)d_in[9];

    const int N = in_sizes[0] / FDIM;     // 50000
    const int E = in_sizes[1] / 2;        // 800000
    const int C = in_sizes[9];            // 40 (post bias length)
    const int* src = ei;
    const int* dst = ei + E;

    // workspace layout
    char* ws = (char*)d_ws;
    size_t off = 0;
    auto carve = [&](size_t bytes) { void* p = ws + off; off = align_up(off + bytes, 256); return p; };
    __bf16* wpre  = (__bf16*)carve((size_t)FDIM * FDIM * 2);
    __bf16* wc1   = (__bf16*)carve((size_t)FDIM * FDIM * 2);
    __bf16* wc2   = (__bf16*)carve((size_t)FDIM * FDIM * 2);
    __bf16* wpost = (__bf16*)carve((size_t)FDIM * C * 2);
    int*    deg   = (int*)carve((size_t)N * 4);
    float*  dinv  = (float*)carve((size_t)N * 4);
    float*  buf0  = (float*)carve((size_t)N * FDIM * 4);
    float*  buf1  = (float*)carve((size_t)N * FDIM * 4);
    float*  buf2  = (float*)carve((size_t)N * FDIM * 4);

    const int BLK = 256;
    auto cdiv = [](int a, int b) { return (a + b - 1) / b; };

    // weights -> bf16
    cvt_w_kernel<<<cdiv(FDIM * FDIM, BLK), BLK, 0, stream>>>(pre_W,   wpre,  FDIM * FDIM);
    cvt_w_kernel<<<cdiv(FDIM * FDIM, BLK), BLK, 0, stream>>>(conv1_W, wc1,   FDIM * FDIM);
    cvt_w_kernel<<<cdiv(FDIM * FDIM, BLK), BLK, 0, stream>>>(conv2_W, wc2,   FDIM * FDIM);
    cvt_w_kernel<<<cdiv(FDIM * C,    BLK), BLK, 0, stream>>>(post_W,  wpost, FDIM * C);

    // degrees + symmetric norm
    zero_int_kernel<<<cdiv(N, BLK), BLK, 0, stream>>>(deg, N);
    edge_degree_kernel<<<cdiv(E, BLK), BLK, 0, stream>>>(dst, deg, E);
    rsqrt_kernel<<<cdiv(N, BLK), BLK, 0, stream>>>(deg, dinv, N);

    const int tiles_m = cdiv(N, 16);
    const int gemm_blocks = cdiv(tiles_m, 8);
    const int nodeThreads = N * 32;                  // wave-per-node kernels
    const int edgeThreads = E * 32;                  // wave-per-edge kernels

    // pre MLP: h = x @ pre_W + pre_b
    gcn_gemm_wmma<8><<<gemm_blocks, BLK, 0, stream>>>(x, wpre, pre_b, buf0, N, FDIM);

    // conv1: h1 = h @ W1 ; agg = A_norm h1 ; relu(agg + b1)
    gcn_gemm_wmma<8><<<gemm_blocks, BLK, 0, stream>>>(buf0, wc1, nullptr, buf1, N, FDIM);
    self_init_kernel<<<cdiv(nodeThreads, BLK), BLK, 0, stream>>>(buf1, dinv, buf2, N);
    edge_scatter_kernel<<<cdiv(edgeThreads, BLK), BLK, 0, stream>>>(buf1, src, dst, dinv, buf2, E);
    bias_relu_kernel<<<cdiv(nodeThreads, BLK), BLK, 0, stream>>>(buf2, conv1_b, N);

    // conv2
    gcn_gemm_wmma<8><<<gemm_blocks, BLK, 0, stream>>>(buf2, wc2, nullptr, buf0, N, FDIM);
    self_init_kernel<<<cdiv(nodeThreads, BLK), BLK, 0, stream>>>(buf0, dinv, buf1, N);
    edge_scatter_kernel<<<cdiv(edgeThreads, BLK), BLK, 0, stream>>>(buf0, src, dst, dinv, buf1, E);
    bias_relu_kernel<<<cdiv(nodeThreads, BLK), BLK, 0, stream>>>(buf1, conv2_b, N);

    // post MLP -> d_out [N x 40]
    gcn_gemm_wmma<3><<<gemm_blocks, BLK, 0, stream>>>(buf1, wpost, post_b, (float*)d_out, N, C);
}
